// VariationalGCNEncoder_40011915329939
// MI455X (gfx1250) — matrix operand order repaired and checked by reading, not compiled, not run on hardware
//
#include <hip/hip_runtime.h>

// Variational GCN encoder for MI455X (gfx1250, wave32).
// GEMMs via v_wmma_f32_16x16x32_bf16 on pre-packed bf16 operands: all hot-loop
// loads are b128 with immediate offsets, software-pipelined one K-chunk ahead
// so WMMAs overlap L2 load latency. GEMM epilogue fuses the self-loop + bias
// init of the aggregation buffer. Edge aggregation via per-edge fp32 atomic
// scatter (entire working set is L2-resident on the 192 MB L2).

typedef __attribute__((ext_vector_type(16))) __bf16 bf16x16;
typedef __attribute__((ext_vector_type(8)))  __bf16 bf16x8;
typedef __attribute__((ext_vector_type(8)))  float  f32x8;

#define NODES 50000
#define EDGES 400000

// ---------------------------------------------------------------------------
// Degree / normalization kernels
// ---------------------------------------------------------------------------
__global__ void deg_init(float* __restrict__ deg, int n) {
    int i = blockIdx.x * blockDim.x + threadIdx.x;
    if (i < n) deg[i] = 1.0f;                       // self loop contributes 1
}

__global__ void deg_accum(const int* __restrict__ dst, float* __restrict__ deg, int e) {
    int i = blockIdx.x * blockDim.x + threadIdx.x;
    if (i < e) atomicAdd(&deg[dst[i]], 1.0f);
}

__global__ void deg_rsqrt_inplace(float* __restrict__ d, int n) {
    int i = blockIdx.x * blockDim.x + threadIdx.x;
    if (i < n) d[i] = rsqrtf(d[i]);                 // deg >= 1 always
}

__global__ void edge_norm_k(const int* __restrict__ src, const int* __restrict__ dst,
                            const float* __restrict__ dinv, float* __restrict__ normE, int e) {
    int i = blockIdx.x * blockDim.x + threadIdx.x;
    if (i < e) normE[i] = dinv[src[i]] * dinv[dst[i]];
}

// ---------------------------------------------------------------------------
// Operand packing
// ---------------------------------------------------------------------------
// fp32 [M x K] -> bf16 [M x K32] row-major, zero-padded in K.
__global__ void cvt_pad_bf16(const float* __restrict__ in, __bf16* __restrict__ outb,
                             int M, int K, int K32)
{
    int idx = blockIdx.x * blockDim.x + threadIdx.x;
    if (idx >= M * K32) return;
    int i = idx / K32;
    int k = idx - i * K32;
    outb[idx] = (k < K) ? (__bf16)in[(size_t)i * K + k] : (__bf16)0.0f;
}

// fp32 W[K x N] -> bf16 Wp[(K32/32)][N][32]: per (chunk, column) the 32 K-values
// are contiguous, so each lane's B fragment (16 values) is two 16-byte runs at bytes 16*half and 16*half+32.
__global__ void pack_w_bf16(const float* __restrict__ W, __bf16* __restrict__ Wp,
                            int K, int N, int K32)
{
    int idx = blockIdx.x * blockDim.x + threadIdx.x;   // total = K32 * N
    if (idx >= K32 * N) return;
    int kk = idx & 31;
    int rest = idx >> 5;
    int n = rest % N;
    int c = rest / N;
    int kg = c * 32 + kk;
    Wp[idx] = (kg < K) ? (__bf16)W[(size_t)kg * N + n] : (__bf16)0.0f;
}

// ---------------------------------------------------------------------------
// bf16 WMMA GEMM with fused aggregation-init epilogue.
//   t[M,N]   = A @ W                     (message source for edge scatter)
//   agg[M,N] = t * dinv[m]^2 + bias[n]   (self-loop + bias pre-init)
// One wave computes NT adjacent 16x16 tiles (shared A fragment).
// CH = K/32 chunks and N are compile-time -> fully unrolled, immediate-offset
// loads, software-pipelined one chunk ahead (distinct register sets so the
// scheduler can keep loads in flight during WMMAs).
// ---------------------------------------------------------------------------
template <int NT, int CH, int N>
__global__ void __launch_bounds__(256)
gemm_bf16_wmma(const __bf16* __restrict__ Abf,   // [M x CH*32] bf16, padded
               const __bf16* __restrict__ Wp,    // packed [CH][N][32] bf16
               const float* __restrict__ dinv,
               const float* __restrict__ bias,
               float* __restrict__ outT,
               float* __restrict__ outAgg, int M)
{
    const int K32  = CH * 32;
    const int lane = threadIdx.x & 31;
    const int wave = blockIdx.x * (blockDim.x >> 5) + (threadIdx.x >> 5);
    const int nGroups = N / (16 * NT);
    const int mT = M >> 4;
    if (wave >= mT * nGroups) return;               // wave-uniform: EXEC all-ones
    const int mTile = wave / nGroups;
    const int nGrp  = wave % nGroups;

    const int half = lane >> 4;                     // 0: lanes 0-15, 1: lanes 16-31
    const int l15  = lane & 15;
    const int rowA = mTile * 16 + l15;

    // A fragment chunk c: elems 0..7 = K(c*32 + half*8 ..+7), elems 8..15 = +16.
    // Base + compile-time byte offsets c*64 and c*64+32.
    const __bf16* __restrict__ paBase = Abf + (size_t)rowA * K32 + half * 8;

    // B fragment chunk c, tile t: elems 0..7 = K(c*32 + half*8 ..+7), elems 8..15 = +16.
    const __bf16* __restrict__ pb[NT];
    int ncol[NT];
    #pragma unroll
    for (int t = 0; t < NT; ++t) {
        ncol[t] = nGrp * (16 * NT) + t * 16 + l15;
        pb[t] = Wp + ((size_t)ncol[t] << 5) + (half << 3);
    }

    auto loadA = [&](int c) -> bf16x16 {
        const bf16x8* __restrict__ pa = (const bf16x8*)(paBase + c * 32);
        return __builtin_shufflevector(pa[0], pa[2],
                   0, 1, 2, 3, 4, 5, 6, 7, 8, 9, 10, 11, 12, 13, 14, 15);
    };
    auto loadB = [&](int c, int t) -> bf16x16 {
        const bf16x8* __restrict__ pbc = (const bf16x8*)(pb[t] + c * N * 32);
        return __builtin_shufflevector(pbc[0], pbc[2],
                   0, 1, 2, 3, 4, 5, 6, 7, 8, 9, 10, 11, 12, 13, 14, 15);
    };

    f32x8 zero = {0.f, 0.f, 0.f, 0.f, 0.f, 0.f, 0.f, 0.f};
    f32x8 acc[NT];
    #pragma unroll
    for (int t = 0; t < NT; ++t) acc[t] = zero;

    // Software pipeline: prefetch chunk c+1 while chunk c is in the matrix unit.
    bf16x16 aCur = loadA(0);
    bf16x16 bCur[NT];
    #pragma unroll
    for (int t = 0; t < NT; ++t) bCur[t] = loadB(0, t);

    #pragma unroll
    for (int c = 0; c < CH; ++c) {
        bf16x16 aNext = aCur;
        bf16x16 bNext[NT];
        #pragma unroll
        for (int t = 0; t < NT; ++t) bNext[t] = bCur[t];
        if (c + 1 < CH) {
            aNext = loadA(c + 1);
            #pragma unroll
            for (int t = 0; t < NT; ++t) bNext[t] = loadB(c + 1, t);
        }
        #pragma unroll
        for (int t = 0; t < NT; ++t)
            acc[t] = __builtin_amdgcn_wmma_f32_16x16x32_bf16(
                         false, aCur, false, bCur[t], (short)0, acc[t], false, false);
        aCur = aNext;
        #pragma unroll
        for (int t = 0; t < NT; ++t) bCur[t] = bNext[t];
    }

    // Epilogue. C/D layout: VGPR r -> row (r + 8*half), col = l15.
    float dv2[8];
    #pragma unroll
    for (int r = 0; r < 8; ++r) {
        float d = dinv[mTile * 16 + r + half * 8];
        dv2[r] = d * d;
    }
    #pragma unroll
    for (int t = 0; t < NT; ++t) {
        float bn = bias[ncol[t]];
        #pragma unroll
        for (int r = 0; r < 8; ++r) {
            size_t o = (size_t)(mTile * 16 + r + half * 8) * N + ncol[t];
            float v = acc[t][r];
            outT[o]   = v;
            outAgg[o] = v * dv2[r] + bn;
        }
    }
}

// ---------------------------------------------------------------------------
// Edge scatter: one wave per edge; lanes stride the feature dimension.
// ---------------------------------------------------------------------------
__global__ void agg_edges(const float* __restrict__ t, const int* __restrict__ src,
                          const int* __restrict__ dst, const float* __restrict__ normE,
                          float* __restrict__ out, int E, int F)
{
    int e    = blockIdx.x * (blockDim.x >> 5) + (threadIdx.x >> 5);
    int lane = threadIdx.x & 31;
    if (e >= E) return;
    int   s = src[e];
    int   d = dst[e];
    float w = normE[e];
    const float* __restrict__ ts = t + (size_t)s * F;
    float* od = out + (size_t)d * F;
    for (int f = lane; f < F; f += 32)
        atomicAdd(&od[f], ts[f] * w);
}

// ReLU fused with bf16 conversion: produces the next GEMM's A operand.
__global__ void relu_cvt_bf16(const float* __restrict__ in, __bf16* __restrict__ outb, int n)
{
    int i = blockIdx.x * blockDim.x + threadIdx.x;
    if (i < n) outb[i] = (__bf16)fmaxf(in[i], 0.0f);
}

// ---------------------------------------------------------------------------
// Launch
// ---------------------------------------------------------------------------
extern "C" void kernel_launch(void* const* d_in, const int* in_sizes, int n_in,
                              void* d_out, int out_size, void* d_ws, size_t ws_size,
                              hipStream_t stream)
{
    const float* x   = (const float*)d_in[0];
    const int*   ei  = (const int*)  d_in[1];
    const float* W1  = (const float*)d_in[2];
    const float* b1  = (const float*)d_in[3];
    const float* W2  = (const float*)d_in[4];
    const float* b2  = (const float*)d_in[5];
    const float* Wmu = (const float*)d_in[6];
    const float* bmu = (const float*)d_in[7];
    const float* Wls = (const float*)d_in[8];
    const float* bls = (const float*)d_in[9];
    float* outp = (float*)d_out;

    const int M = NODES, E = EDGES;
    const int K1 = 109, F1 = 320, F2 = 64, F3 = 32;

    const int* src = ei;          // edge_index[0]
    const int* dst = ei + E;      // edge_index[1]

    // Workspace: dinv | normE | bufT (fp32) | bufAgg (fp32) | bfIn (bf16) | Wp (bf16)
    float*  dinv   = (float*)d_ws;
    float*  normE  = dinv + 50048;
    float*  bufT   = normE + 400000;                      // M*F1 fp32 (64 MB)
    float*  bufAgg = bufT + (size_t)M * F1;               // M*F1 fp32 (64 MB)
    __bf16* bfIn   = (__bf16*)(bufAgg + (size_t)M * F1);  // M*F1 bf16 (32 MB; x/h1/h2)
    __bf16* Wp     = bfIn + (size_t)M * F1;               // 128*320 bf16 max (80 KB)

    const int TPB = 256;

    // --- degrees + per-edge norms (computed once, reused by all 4 convs) ---
    deg_init         <<<(M + TPB - 1) / TPB, TPB, 0, stream>>>(dinv, M);
    deg_accum        <<<(E + TPB - 1) / TPB, TPB, 0, stream>>>(dst, dinv, E);
    deg_rsqrt_inplace<<<(M + TPB - 1) / TPB, TPB, 0, stream>>>(dinv, M);
    edge_norm_k      <<<(E + TPB - 1) / TPB, TPB, 0, stream>>>(src, dst, dinv, normE, E);

    // --- layer 1: GCNConv(109 -> 320) + ReLU ---
    {
        int nck = M * 128;
        cvt_pad_bf16<<<(nck + TPB - 1) / TPB, TPB, 0, stream>>>(x, bfIn, M, K1, 128);
        int nwp = 128 * F1;
        pack_w_bf16<<<(nwp + TPB - 1) / TPB, TPB, 0, stream>>>(W1, Wp, K1, F1, 128);
        int waves = (M / 16) * (F1 / 64);            // NT=4
        gemm_bf16_wmma<4, 4, 320><<<(waves + 7) / 8, TPB, 0, stream>>>(
            bfIn, Wp, dinv, b1, bufT, bufAgg, M);
        agg_edges<<<(E + 7) / 8, TPB, 0, stream>>>(bufT, src, dst, normE, bufAgg, E, F1);
        int nf = M * F1;
        relu_cvt_bf16<<<(nf + TPB - 1) / TPB, TPB, 0, stream>>>(bufAgg, bfIn, nf);
    }

    // --- layer 2: GCNConv(320 -> 64) + ReLU ---
    {
        int nwp = 320 * F2;
        pack_w_bf16<<<(nwp + TPB - 1) / TPB, TPB, 0, stream>>>(W2, Wp, F1, F2, 320);
        int waves = (M / 16) * (F2 / 64);            // NT=4
        gemm_bf16_wmma<4, 10, 64><<<(waves + 7) / 8, TPB, 0, stream>>>(
            bfIn, Wp, dinv, b2, bufT, bufAgg, M);
        agg_edges<<<(E + 7) / 8, TPB, 0, stream>>>(bufT, src, dst, normE, bufAgg, E, F2);
        int nf = M * F2;
        relu_cvt_bf16<<<(nf + TPB - 1) / TPB, TPB, 0, stream>>>(bufAgg, bfIn, nf);
    }

    // --- layer 3: GCNConv(64 -> 32) -> mu (first half of d_out) ---
    {
        int nwp = 64 * F3;
        pack_w_bf16<<<(nwp + TPB - 1) / TPB, TPB, 0, stream>>>(Wmu, Wp, F2, F3, 64);
        int waves = (M / 16) * (F3 / 32);            // NT=2
        gemm_bf16_wmma<2, 2, 32><<<(waves + 7) / 8, TPB, 0, stream>>>(
            bfIn, Wp, dinv, bmu, bufT, outp, M);
        agg_edges<<<(E + 7) / 8, TPB, 0, stream>>>(bufT, src, dst, normE, outp, E, F3);
    }

    // --- layer 4: GCNConv(64 -> 32) -> logstd (second half of d_out) ---
    {
        int nwp = 64 * F3;
        pack_w_bf16<<<(nwp + TPB - 1) / TPB, TPB, 0, stream>>>(Wls, Wp, F2, F3, 64);
        int waves = (M / 16) * (F3 / 32);            // NT=2
        float* outLs = outp + (size_t)M * F3;
        gemm_bf16_wmma<2, 2, 32><<<(waves + 7) / 8, TPB, 0, stream>>>(
            bfIn, Wp, dinv, bls, bufT, outLs, M);
        agg_edges<<<(E + 7) / 8, TPB, 0, stream>>>(bufT, src, dst, normE, outLs, E, F3);
    }
}
